// TransformerEncoder_43877385896501
// MI455X (gfx1250) — matrix operand order repaired and checked by reading, not compiled, not run on hardware
//
#include <hip/hip_runtime.h>

typedef _Float16 v16h __attribute__((ext_vector_type(16)));
typedef _Float16 v8h  __attribute__((ext_vector_type(8)));
typedef float    v8f  __attribute__((ext_vector_type(8)));

#define BB 16
#define SS 1024
#define DD 256
#define HH 8
#define DH 32
#define LL 4
#define OBS 160
#define ACT 96
#define MS  (BB*SS)        // 16384 rows

static __device__ __forceinline__ v8f wmma16x16x32(v16h a, v16h b, v8f c) {
    return __builtin_amdgcn_wmma_f32_16x16x32_f16(false, a, false, b, (short)0, c, false, false);
}

// ---------------------------------------------------------------------------
// f32 -> f16 convert (weights only; activation converts are fused away)
// ---------------------------------------------------------------------------
__global__ void cvt_f32_f16(const float* __restrict__ src, _Float16* __restrict__ dst, int n) {
    int i = blockIdx.x * blockDim.x + threadIdx.x;
    if (i < n) dst[i] = (_Float16)src[i];
}

// ---------------------------------------------------------------------------
// concat(obs, act) -> f16 A matrix (MS x 256)
// ---------------------------------------------------------------------------
__global__ void concat_obs_act(const float* __restrict__ obs, const float* __restrict__ act,
                               _Float16* __restrict__ X) {
    int i = blockIdx.x * blockDim.x + threadIdx.x;  // over MS*256
    if (i >= MS * DD) return;
    int col = i & 255;
    int row = i >> 8;
    float v = (col < OBS) ? obs[(size_t)row * OBS + col]
                          : act[(size_t)row * ACT + (col - OBS)];
    X[i] = (_Float16)v;
}

// ---------------------------------------------------------------------------
// WMMA GEMM: C[M,256] = A[M,256](f16) @ B[256,256](f16) + bias (+ res)
//   - B panel (256 x 64 f16 = 32 KB) staged in LDS via CDNA5 async copy
//     (global_load_async_to_lds_b128 / ASYNCcnt), reused by all 4 waves
//     and all 8 K-steps.
//   - A row preloaded to registers; B tiles double-buffered across K-steps
//     so the wmma quartet never waits on the loads issued in the same step.
//   - Optional f16 side-output Ch feeds the next GEMM's A operand.
// wave computes 16x64; block = 4 waves = 64 rows; gridDim.y*64 covers N=256.
// ---------------------------------------------------------------------------
__global__ __launch_bounds__(128) void gemm_f16_wmma(
    const _Float16* __restrict__ A, const _Float16* __restrict__ Bw,
    const float* __restrict__ bias, const float* __restrict__ res,
    float* __restrict__ C, _Float16* __restrict__ Ch)
{
    __shared__ __align__(32) _Float16 Bls[DD * 64];   // [k][64 n] row-major, 32 KB

    const int lane  = threadIdx.x & 31;
    const int wave  = threadIdx.x >> 5;
    const int mBase = blockIdx.x * 64 + wave * 16;
    const int nBase = blockIdx.y * 64;
    const int half  = lane >> 4;
    const int lm    = lane & 15;

    // ---- async-stage B panel into LDS: 32 KB = 128 thr * 16 iters * 16 B ----
    {
        unsigned ldsBase = (unsigned)(size_t)&Bls[0];       // low 32 bits = LDS offset
        const char* gbase = (const char*)Bw + (size_t)nBase * 2;
        #pragma unroll
        for (int it = 0; it < 16; ++it) {
            unsigned L = (unsigned)threadIdx.x * 16u + (unsigned)it * 2048u; // byte off in panel
            unsigned krow = L >> 7;                         // 128 B per 64-half row
            unsigned cb   = L & 127u;
            unsigned long long ga =
                (unsigned long long)(size_t)(gbase + (size_t)krow * 512 + cb);
            asm volatile("global_load_async_to_lds_b128 %0, %1, off"
                         :: "v"(ldsBase + L), "v"(ga) : "memory");
        }
        asm volatile("s_wait_asynccnt 0x0" ::: "memory");
        __syncthreads();
    }

    // ---- preload A row (ISA 16-bit A layout chunks) ----
    const _Float16* Arow = A + (size_t)(mBase + lm) * DD;
    v8h areg[16];
    #pragma unroll
    for (int ks = 0; ks < 8; ++ks) {
        areg[2 * ks]     = *(const v8h*)(Arow + ks * 32 + 8 * half);
        areg[2 * ks + 1] = *(const v8h*)(Arow + ks * 32 + 16 + 8 * half);
    }

    v8f acc[4] = {v8f{}, v8f{}, v8f{}, v8f{}};

    // prologue: B tiles for K-step 0
    v16h b0, b1, b2, b3;
    {
        const _Float16* Brow = &Bls[lm];
        #pragma unroll
        for (int i = 0; i < 8; ++i) {
            const int k0 = (8 * half + i) * 64, k1 = (16 + 8 * half + i) * 64;
            b0[i] = Brow[k0];      b0[8 + i] = Brow[k1];
            b1[i] = Brow[k0 + 16]; b1[8 + i] = Brow[k1 + 16];
            b2[i] = Brow[k0 + 32]; b2[8 + i] = Brow[k1 + 32];
            b3[i] = Brow[k0 + 48]; b3[8 + i] = Brow[k1 + 48];
        }
    }

    #pragma unroll
    for (int ks = 0; ks < 8; ++ks) {
        v16h a;
        #pragma unroll
        for (int i = 0; i < 8; ++i) { a[i] = areg[2 * ks][i]; a[8 + i] = areg[2 * ks + 1][i]; }

        // prefetch next K-step's B tiles (double buffer)
        v16h n0, n1, n2, n3;
        if (ks < 7) {
            const _Float16* Bn = &Bls[(ks + 1) * 32 * 64 + lm];
            #pragma unroll
            for (int i = 0; i < 8; ++i) {
                const int k0 = (8 * half + i) * 64, k1 = (16 + 8 * half + i) * 64;
                n0[i] = Bn[k0];      n0[8 + i] = Bn[k1];
                n1[i] = Bn[k0 + 16]; n1[8 + i] = Bn[k1 + 16];
                n2[i] = Bn[k0 + 32]; n2[8 + i] = Bn[k1 + 32];
                n3[i] = Bn[k0 + 48]; n3[8 + i] = Bn[k1 + 48];
            }
        }

        acc[0] = wmma16x16x32(a, b0, acc[0]);
        acc[1] = wmma16x16x32(a, b1, acc[1]);
        acc[2] = wmma16x16x32(a, b2, acc[2]);
        acc[3] = wmma16x16x32(a, b3, acc[3]);

        if (ks < 7) { b0 = n0; b1 = n1; b2 = n2; b3 = n3; }
    }

    #pragma unroll
    for (int t = 0; t < 4; ++t) {
        const int n  = nBase + t * 16 + lm;
        const float bv = bias[n];
        #pragma unroll
        for (int r = 0; r < 8; ++r) {
            const int row = mBase + r + 8 * half;  // C/D layout: m = r + 8*(lane/16)
            const size_t idx = (size_t)row * DD + n;
            float v = acc[t][r] + bv;
            if (res) v += res[idx];
            C[idx] = v;
            if (Ch) Ch[idx] = (_Float16)v;
        }
    }
}

// ---------------------------------------------------------------------------
// RoPE + head packing.  Reads fp32 Q/K/V proj (B,S,D); writes
//   Qh  [B,H,S,DH] f16   (A-matrix source)
//   Kth [B,H,DH,S] f16   (pre-transposed so score B-tiles load contiguously)
//   Vh  [B,H,S,DH] f16   (PV B-tiles load contiguously)
// ---------------------------------------------------------------------------
__global__ void rope_pack(const float* __restrict__ qf, const float* __restrict__ kf,
                          const float* __restrict__ vf,
                          _Float16* __restrict__ Qh, _Float16* __restrict__ Kth,
                          _Float16* __restrict__ Vh)
{
    int idx = blockIdx.x * blockDim.x + threadIdx.x;  // over B*S*H
    if (idx >= BB * SS * HH) return;
    const int h = idx & 7;
    const int s = (idx >> 3) & (SS - 1);
    const int b = idx >> 13;
    const size_t rowoff = ((size_t)(b * SS + s)) * DD + h * DH;
    const size_t bh = (size_t)(b * HH + h);
    _Float16* qd = Qh  + (bh * SS + s) * DH;
    _Float16* vd = Vh  + (bh * SS + s) * DH;
    _Float16* kd = Kth + bh * DH * SS + s;   // stride SS per dh element

    #pragma unroll
    for (int i = 0; i < DH / 2; ++i) {
        // inv_freq = 10000^(-2i/32) = exp(-i * ln(10000)/16)
        float inv = __expf(-(float)i * 0.57564627325f);
        float th  = (float)s * inv;
        float sn, cs;
        __sincosf(th, &sn, &cs);
        float qr = qf[rowoff + 2 * i], qi = qf[rowoff + 2 * i + 1];
        qd[2 * i]     = (_Float16)(qr * cs - qi * sn);
        qd[2 * i + 1] = (_Float16)(qr * sn + qi * cs);
        float kr = kf[rowoff + 2 * i], ki = kf[rowoff + 2 * i + 1];
        kd[(2 * i) * SS]     = (_Float16)(kr * cs - ki * sn);
        kd[(2 * i + 1) * SS] = (_Float16)(kr * sn + ki * cs);
        vd[2 * i]     = (_Float16)vf[rowoff + 2 * i];
        vd[2 * i + 1] = (_Float16)vf[rowoff + 2 * i + 1];
    }
}

// ---------------------------------------------------------------------------
// Flash-style attention: wave = one (b,h, 16-query block).
// Per 32-key block: 2 wmma (scores vs Kt), online softmax, LDS relayout of P,
// 2 wmma (P @ V). K/V tile loads are hoisted above the softmax VALU work so
// their latency hides under exp/shfl. Output written in f16 (O-proj A input).
// ---------------------------------------------------------------------------
__global__ __launch_bounds__(128) void attn_wmma(
    const _Float16* __restrict__ Q, const _Float16* __restrict__ Kt,
    const _Float16* __restrict__ V, _Float16* __restrict__ O)
{
    __shared__ __align__(32) _Float16 Pls[4][16][32];   // per-wave P tile
    const int lane = threadIdx.x & 31;
    const int wave = threadIdx.x >> 5;
    const int gw   = blockIdx.x * 4 + wave;            // [0, B*H*64)
    const int b    = gw >> 9;
    const int h    = (gw >> 6) & 7;
    const int q0   = (gw & 63) * 16;
    const int half = lane >> 4;
    const int lm   = lane & 15;

    const size_t bh = (size_t)(b * HH + h);
    const _Float16* Qp  = Q  + (bh * SS + q0) * DH;
    const _Float16* Ktp = Kt + bh * DH * SS;
    const _Float16* Vp  = V  + bh * SS * DH;

    // Q tile (A layout), DH==32 == one K step
    v8h qlo = *(const v8h*)(Qp + (size_t)lm * DH + 8 * half);
    v8h qhi = *(const v8h*)(Qp + (size_t)lm * DH + 16 + 8 * half);
    v16h aQ;
    #pragma unroll
    for (int i = 0; i < 8; ++i) { aQ[i] = qlo[i]; aQ[8 + i] = qhi[i]; }

    float rmax[8], rsum[8];
    #pragma unroll
    for (int r = 0; r < 8; ++r) { rmax[r] = -1.0e30f; rsum[r] = 0.0f; }
    v8f acc0{}, acc1{};
    const float scale = 0.17677669529663689f;   // 1/sqrt(32)

    // prologue: Kt tiles for key block 0
    v16h kt0, kt1;
    #pragma unroll
    for (int i = 0; i < 8; ++i) {
        kt0[i] = Ktp[(size_t)(8 * half + i) * SS + lm];      kt0[8 + i] = Ktp[(size_t)(16 + 8 * half + i) * SS + lm];
        kt1[i] = Ktp[(size_t)(8 * half + i) * SS + 16 + lm]; kt1[8 + i] = Ktp[(size_t)(16 + 8 * half + i) * SS + 16 + lm];
    }

    for (int kb = 0; kb < SS; kb += 32) {
        v8f s0{}, s1{};
        s0 = wmma16x16x32(aQ, kt0, s0);
        s1 = wmma16x16x32(aQ, kt1, s1);

        // hoisted loads: V tiles of this block + Kt tiles of the next block
        v16h v0, v1;
        #pragma unroll
        for (int i = 0; i < 8; ++i) {
            v0[i] = Vp[(size_t)(kb + 8 * half + i) * DH + lm];      v0[8 + i] = Vp[(size_t)(kb + 16 + 8 * half + i) * DH + lm];
            v1[i] = Vp[(size_t)(kb + 8 * half + i) * DH + 16 + lm]; v1[8 + i] = Vp[(size_t)(kb + 16 + 8 * half + i) * DH + 16 + lm];
        }
        if (kb + 32 < SS) {
            #pragma unroll
            for (int i = 0; i < 8; ++i) {
                kt0[i] = Ktp[(size_t)(8 * half + i) * SS + kb + 32 + lm]; kt0[8 + i] = Ktp[(size_t)(16 + 8 * half + i) * SS + kb + 32 + lm];
                kt1[i] = Ktp[(size_t)(8 * half + i) * SS + kb + 48 + lm]; kt1[8 + i] = Ktp[(size_t)(16 + 8 * half + i) * SS + kb + 48 + lm];
            }
        }

        #pragma unroll
        for (int r = 0; r < 8; ++r) {
            float x0 = s0[r] * scale, x1 = s1[r] * scale;
            float mx = fmaxf(x0, x1);
            #pragma unroll
            for (int off = 1; off < 16; off <<= 1) mx = fmaxf(mx, __shfl_xor(mx, off, 32));
            float nm = fmaxf(rmax[r], mx);
            float f  = __expf(rmax[r] - nm);
            float p0 = __expf(x0 - nm), p1 = __expf(x1 - nm);
            float ts = p0 + p1;
            #pragma unroll
            for (int off = 1; off < 16; off <<= 1) ts += __shfl_xor(ts, off, 32);
            rsum[r] = rsum[r] * f + ts;
            rmax[r] = nm;
            acc0[r] *= f;
            acc1[r] *= f;
            const int m = r + 8 * half;
            Pls[wave][m][lm]      = (_Float16)p0;
            Pls[wave][m][16 + lm] = (_Float16)p1;
        }
        // CDNA5 split-counter LDS sync (same-wave cross-lane relayout)
        asm volatile("s_wait_dscnt 0x0" ::: "memory");

        // reload P in A layout
        v8h plo = *(const v8h*)(&Pls[wave][lm][8 * half]);
        v8h phi = *(const v8h*)(&Pls[wave][lm][16 + 8 * half]);
        v16h aP;
        #pragma unroll
        for (int i = 0; i < 8; ++i) { aP[i] = plo[i]; aP[8 + i] = phi[i]; }

        acc0 = wmma16x16x32(aP, v0, acc0);
        acc1 = wmma16x16x32(aP, v1, acc1);
        asm volatile("s_wait_dscnt 0x0" ::: "memory");
    }

    _Float16* Ob = O + ((size_t)b * SS + q0) * DD + h * DH;
    #pragma unroll
    for (int r = 0; r < 8; ++r) {
        const int m = r + 8 * half;
        const float inv = 1.0f / rsum[r];
        Ob[(size_t)m * DD + lm]      = (_Float16)(acc0[r] * inv);
        Ob[(size_t)m * DD + 16 + lm] = (_Float16)(acc1[r] * inv);
    }
}

// ---------------------------------------------------------------------------
// Final LayerNorm: one block (256 thr) per row; writes seq embeddings and,
// for s == S-1, the context embedding.
// ---------------------------------------------------------------------------
__global__ __launch_bounds__(256) void layernorm_out(
    const float* __restrict__ y, const float* __restrict__ g,
    const float* __restrict__ bt, float* __restrict__ out)
{
    __shared__ float red[256];
    const int row = blockIdx.x;          // 0..MS-1
    const int c   = threadIdx.x;         // 0..255
    const float v = y[(size_t)row * DD + c];

    red[c] = v; __syncthreads();
    #pragma unroll
    for (int s = 128; s > 0; s >>= 1) { if (c < s) red[c] += red[c + s]; __syncthreads(); }
    const float mu = red[0] * (1.0f / DD);
    __syncthreads();
    const float d = v - mu;
    red[c] = d * d; __syncthreads();
    #pragma unroll
    for (int s = 128; s > 0; s >>= 1) { if (c < s) red[c] += red[c + s]; __syncthreads(); }
    const float var = red[0] * (1.0f / DD);

    const float r = d * rsqrtf(var + 1e-5f) * g[c] + bt[c];
    float* seq = out + BB * DD;          // context block first, then sequence
    seq[(size_t)row * DD + c] = r;
    if ((row & (SS - 1)) == SS - 1) out[(row >> 10) * DD + c] = r;
}

// ---------------------------------------------------------------------------
// Host orchestration
// ---------------------------------------------------------------------------
extern "C" void kernel_launch(void* const* d_in, const int* in_sizes, int n_in,
                              void* d_out, int out_size, void* d_ws, size_t ws_size,
                              hipStream_t stream) {
    const float* obs   = (const float*)d_in[0];
    const float* act   = (const float*)d_in[1];
    const float* in_W  = (const float*)d_in[2];
    const float* in_b  = (const float*)d_in[3];
    const float* qW    = (const float*)d_in[4];
    const float* qb    = (const float*)d_in[5];
    const float* kW    = (const float*)d_in[6];
    const float* kb    = (const float*)d_in[7];
    const float* vW    = (const float*)d_in[8];
    const float* vb    = (const float*)d_in[9];
    const float* oW    = (const float*)d_in[10];
    const float* ob    = (const float*)d_in[11];
    const float* out_W = (const float*)d_in[12];
    const float* out_b = (const float*)d_in[13];
    const float* ln_g  = (const float*)d_in[14];
    const float* ln_b  = (const float*)d_in[15];

    size_t off = 0;
    char* base = (char*)d_ws;
    auto carve = [&](size_t bytes) -> void* {
        void* p = base + off;
        off = (off + bytes + 255) & ~(size_t)255;
        return p;
    };
    const size_t actF32 = (size_t)MS * DD * sizeof(float);
    const size_t actF16 = (size_t)MS * DD * sizeof(_Float16);
    float*    x    = (float*)carve(actF32);
    float*    t0   = (float*)carve(actF32);
    float*    t1   = (float*)carve(actF32);
    float*    t2   = (float*)carve(actF32);
    _Float16* ch   = (_Float16*)carve(actF16);   // concat input, f16
    _Float16* xh   = (_Float16*)carve(actF16);   // current x, f16
    _Float16* ah   = (_Float16*)carve(actF16);   // attention out, f16
    _Float16* qh   = (_Float16*)carve(actF16);
    _Float16* kth  = (_Float16*)carve(actF16);
    _Float16* vh   = (_Float16*)carve(actF16);
    _Float16* inWh  = (_Float16*)carve((size_t)DD * DD * 2);
    _Float16* qWh   = (_Float16*)carve((size_t)LL * DD * DD * 2);
    _Float16* kWh   = (_Float16*)carve((size_t)LL * DD * DD * 2);
    _Float16* vWh   = (_Float16*)carve((size_t)LL * DD * DD * 2);
    _Float16* oWh   = (_Float16*)carve((size_t)LL * DD * DD * 2);
    _Float16* outWh = (_Float16*)carve((size_t)DD * DD * 2);

    const int WDD = DD * DD;           // 65536
    cvt_f32_f16<<<(WDD + 255) / 256, 256, 0, stream>>>(in_W,  inWh,  WDD);
    cvt_f32_f16<<<(LL * WDD + 255) / 256, 256, 0, stream>>>(qW, qWh, LL * WDD);
    cvt_f32_f16<<<(LL * WDD + 255) / 256, 256, 0, stream>>>(kW, kWh, LL * WDD);
    cvt_f32_f16<<<(LL * WDD + 255) / 256, 256, 0, stream>>>(vW, vWh, LL * WDD);
    cvt_f32_f16<<<(LL * WDD + 255) / 256, 256, 0, stream>>>(oW, oWh, LL * WDD);
    cvt_f32_f16<<<(WDD + 255) / 256, 256, 0, stream>>>(out_W, outWh, WDD);

    const int NA = MS * DD;            // 4,194,304 activation elements
    dim3 gemmGrid(MS / 64, 4);

    // x = concat(obs, act) @ in_W + in_b   (also emits xh = f16(x))
    concat_obs_act<<<NA / 256, 256, 0, stream>>>(obs, act, ch);
    gemm_f16_wmma<<<gemmGrid, 128, 0, stream>>>(ch, inWh, in_b, nullptr, x, xh);

    for (int l = 0; l < LL; ++l) {
        gemm_f16_wmma<<<gemmGrid, 128, 0, stream>>>(xh, qWh + (size_t)l * WDD, qb + l * DD, nullptr, t0, nullptr);
        gemm_f16_wmma<<<gemmGrid, 128, 0, stream>>>(xh, kWh + (size_t)l * WDD, kb + l * DD, nullptr, t1, nullptr);
        gemm_f16_wmma<<<gemmGrid, 128, 0, stream>>>(xh, vWh + (size_t)l * WDD, vb + l * DD, nullptr, t2, nullptr);
        rope_pack<<<(BB * SS * HH) / 256, 256, 0, stream>>>(t0, t1, t2, qh, kth, vh);
        attn_wmma<<<(BB * HH * (SS / 16)) / 4, 128, 0, stream>>>(qh, kth, vh, ah);
        // x = x + attn_out @ oW + ob ; also emits xh = f16(x') for next layer
        gemm_f16_wmma<<<gemmGrid, 128, 0, stream>>>(ah, oWh + (size_t)l * WDD, ob + l * DD, x, x, xh);
    }

    gemm_f16_wmma<<<gemmGrid, 128, 0, stream>>>(xh, outWh, out_b, nullptr, t0, nullptr);
    layernorm_out<<<MS, 256, 0, stream>>>(t0, ln_g, ln_b, (float*)d_out);
}
